// CompressedLinear_48026324304149
// MI455X (gfx1250) — matrix operand order repaired and checked by reading, not compiled, not run on hardware
//
#include <hip/hip_runtime.h>

// CompressedLinear as an f16-WMMA GEMM on gfx1250 (wave32).
//   M = B*S = 8192, N = OUT = 11008, K = IN = 4096
//   out = (x_f16 @ codes_f16^T) * scale[n] + bias[n]   (scale factored out of K-sum)
// Block tile 128(M) x 256(N), 8 waves in 2x4 grid, wave tile 64x64:
//   16 WMMAs vs 16 LDS fragment loads per K-step (4x reuse on A and B frags).

typedef __attribute__((ext_vector_type(16))) _Float16 v16h;
typedef __attribute__((ext_vector_type(8)))  _Float16 v8h;
typedef __attribute__((ext_vector_type(8)))  float    v8f;
typedef __attribute__((ext_vector_type(4)))  float    v4f;
typedef __attribute__((ext_vector_type(4)))  int      v4i;

#define M_TOT 8192
#define K_TOT 4096
#define N_TOT 11008
#define MT 128            // block tile M
#define NT 256            // block tile N
#define KT 32             // K step (one wmma_f32_16x16x32_f16 per acc tile per step)
#define LDH 40            // LDS row stride in halfs (80B: 16B-aligned, bank-spread)
#define NSTEPS (K_TOT / KT)

#define CAT16(lo, hi8) __builtin_shufflevector((lo), (hi8), \
    0, 1, 2, 3, 4, 5, 6, 7, 8, 9, 10, 11, 12, 13, 14, 15)

__global__ __launch_bounds__(256)
void compressed_linear_wmma_f16(const float* __restrict__ x,
                                const int*   __restrict__ w8,
                                const float* __restrict__ scale,
                                const float* __restrict__ bias,
                                float* __restrict__ out) {
  // Double-buffered tiles: A = x (MT x KT), B = weight rows (NT x KT), f16.
  __shared__ _Float16 lds_a[2][MT * LDH];   // 2 * 10240 B
  __shared__ _Float16 lds_b[2][NT * LDH];   // 2 * 20480 B   (total 60 KB)

  const int tid  = threadIdx.x;
  const int lane = tid & 31;       // wave32 lane
  const int wv   = tid >> 5;       // 8 waves per block
  const int lr   = lane & 15;      // row-within-fragment lane
  const int hi   = lane >> 4;      // half-wave selector (K-split per ISA layout)
  const int wm   = (wv >> 2) * 64; // wave M offset (2 waves in M)
  const int wn   = (wv & 3)  * 64; // wave N offset (4 waves in N)

  const int m_tile = blockIdx.y * MT;
  const int n_tile = blockIdx.x * NT;

  // Cooperative loaders:
  //   A: each thread owns 16 consecutive K of one of 128 rows (2 threads/row).
  //   B: each thread owns one full 32-K row of the 256-row weight tile.
  const int a_row = tid >> 1;              // 0..127
  const int a_col = (tid & 1) << 4;        // 0 or 16

  const float* xg = x  + (size_t)(m_tile + a_row) * K_TOT + a_col;
  const int*   wg = w8 + (size_t)(n_tile + tid) * K_TOT;

  v8f acc[4][4] = {};

  auto load_tile = [&](int ks, int buf) {
    const float* xa = xg + (size_t)ks * KT;
    const int*   wa = wg + (size_t)ks * KT;
    // A: 16 floats -> 16 halfs
    v8h ha[2];
#pragma unroll
    for (int c = 0; c < 2; ++c) {
      v4f x0 = *(const v4f*)(xa + c * 8);
      v4f x1 = *(const v4f*)(xa + c * 8 + 4);
#pragma unroll
      for (int e = 0; e < 4; ++e) {
        ha[c][e]     = (_Float16)x0[e];
        ha[c][e + 4] = (_Float16)x1[e];
      }
    }
    // B: 32 int codes -> 32 halfs (codes 0..126 are exact in f16)
    v8h hb[4];
#pragma unroll
    for (int c = 0; c < 4; ++c) {
      v4i w0 = *(const v4i*)(wa + c * 8);
      v4i w1 = *(const v4i*)(wa + c * 8 + 4);
#pragma unroll
      for (int e = 0; e < 4; ++e) {
        hb[c][e]     = (_Float16)w0[e];
        hb[c][e + 4] = (_Float16)w1[e];
      }
    }
    _Float16* la = &lds_a[buf][a_row * LDH + a_col];
    *(v8h*)(la)     = ha[0];
    *(v8h*)(la + 8) = ha[1];
    _Float16* lb = &lds_b[buf][tid * LDH];
#pragma unroll
    for (int c = 0; c < 4; ++c)
      *(v8h*)(lb + c * 8) = hb[c];
  };

  load_tile(0, 0);
  __syncthreads();

  for (int ks = 0; ks < NSTEPS; ++ks) {
    const int cur = ks & 1;
    if (ks + 1 < NSTEPS) load_tile(ks + 1, cur ^ 1);
    if (ks + 2 < NSTEPS) {
      __builtin_prefetch(xg + (size_t)(ks + 2) * KT, 0, 1);
      __builtin_prefetch(wg + (size_t)(ks + 2) * KT, 0, 1);
    }

    // B fragments (32x16, K x N): lane holds column n = wn + j*16 + lr,
    // 16 contiguous K at offset hi*16 (lanes 16-31 take K=16..31).
    v16h bf[4];
#pragma unroll
    for (int j = 0; j < 4; ++j) {
      const _Float16* p = &lds_b[cur][(wn + j * 16 + lr) * LDH + hi * 16];
      v8h lo  = *(const v8h*)(p);
      v8h hi8 = *(const v8h*)(p + 8);
      bf[j] = CAT16(lo, hi8);
    }
    // A fragments (16x32, M x K): lane holds row m = wm + i*16 + lr,
    // 16 contiguous K at offset hi*16 (matches B fragment K order).
    v16h af[4];
#pragma unroll
    for (int i = 0; i < 4; ++i) {
      const _Float16* p = &lds_a[cur][(wm + i * 16 + lr) * LDH + hi * 16];
      v8h lo  = *(const v8h*)(p);
      v8h hi8 = *(const v8h*)(p + 8);
      af[i] = CAT16(lo, hi8);
    }

#pragma unroll
    for (int i = 0; i < 4; ++i)
#pragma unroll
      for (int j = 0; j < 4; ++j)
        acc[i][j] = __builtin_amdgcn_wmma_f32_16x16x32_f16(
            false, af[i], false, bf[j], (short)0, acc[i][j], false, false);

    __syncthreads();
  }

  // Epilogue: fp32 scale + bias, C/D layout m = v + 8*hi, n = lane&15.
#pragma unroll
  for (int j = 0; j < 4; ++j) {
    const int   n  = n_tile + wn + j * 16 + lr;
    const float s  = scale[n];
    const float bb = bias[n];
#pragma unroll
    for (int i = 0; i < 4; ++i) {
      const int m0 = m_tile + wm + i * 16 + hi * 8;
      float* o = out + (size_t)m0 * N_TOT + n;
#pragma unroll
      for (int v = 0; v < 8; ++v)
        o[(size_t)v * N_TOT] = acc[i][j][v] * s + bb;
    }
  }
}

extern "C" void kernel_launch(void* const* d_in, const int* in_sizes, int n_in,
                              void* d_out, int out_size, void* d_ws, size_t ws_size,
                              hipStream_t stream) {
  const float* x     = (const float*)d_in[0];   // [4, 2048, 4096] fp32
  const int*   w8    = (const int*)d_in[1];     // [11008, 4096] int32 codes
  const float* scale = (const float*)d_in[2];   // [11008] fp32
  const float* bias  = (const float*)d_in[3];   // [11008] fp32
  float*       out   = (float*)d_out;           // [4, 2048, 11008] fp32

  dim3 grid(N_TOT / NT, M_TOT / MT);  // (43, 64)
  dim3 block(256);                    // 8 wave32 waves
  compressed_linear_wmma_f16<<<grid, block, 0, stream>>>(x, w8, scale, bias, out);
}
